// HyperPatchInvertedResidual_32375463477392
// MI455X (gfx1250) — matrix operand order repaired and checked by reading, not compiled, not run on hardware
//
#include <hip/hip_runtime.h>
#include <hip/hip_bf16.h>
#include <math.h>

// ---------------------------------------------------------------------------
// HyperPatchInvertedResidual for MI455X (gfx1250, wave32, WMMA)
//   x : (8, 32, 128, 128) f32       s : (8, 14016, 8, 8) f32
//   512 patches of 16x16 (18x18 reflect halo), per-patch weights from s.
//   Stage1: GEMM 324x32 @ 32x192   (bf16 WMMA 16x16x32, f32 accum)
//   Stage2: depthwise 3x3           (VALU, fused BN1+ReLU6)
//   Stage3: GEMM 256x192 @ 192x32   (bf16 WMMA, 6-step K accumulation)
//   Global batchnorm between stages -> multi-pass, atomic stat reduction.
//   Intermediates bf16 channel-major (b128 stores, coalesced reads),
//   LDS zero-padded so WMMA loops are fully branch-free, fragments load
//   as ds_load_b128 pairs, wave id routed through readfirstlane so tile
//   loops use scalar branches (EXEC stays all-ones around WMMA).
// ---------------------------------------------------------------------------

typedef __bf16 bf16_t;
typedef __attribute__((ext_vector_type(16))) __bf16   v16bf;
typedef __attribute__((ext_vector_type(8)))  float    v8f;
typedef __attribute__((ext_vector_type(8)))  unsigned v8u;
typedef __attribute__((ext_vector_type(4)))  unsigned uint4v;
typedef __attribute__((ext_vector_type(4)))  float    float4v;

#define IN_NC   32
#define OUT_NC  32
#define HIDDEN  192
#define NPATCH  512
#define HYPER   14016
#define R1      6144         // offset of w2 block in wt
#define R2      7872         // offset of w3 block in wt
#define M1      324          // 18*18 halo positions
#define M1P     336          // padded to 21*16: branch-free WMMA epilogue
#define M3      256          // 16*16 outputs

union Frag  { v8u u; uint4v u4[2]; v16bf h; }; // WMMA operand: 2 x b128 loads
union Pack8 { bf16_t h[8]; uint4v u4; };       // 8 bf16 -> one b128 store
union Acc8  { v8f a; float4v q[2]; };          // 8 f32  -> two b128 stores
union Pair  { bf16_t h[2]; unsigned u; };      // bf16 pair pack

// per-patch hyper-weight fetch: wt[p, q] = s[b, q, fy, fx]
__device__ __forceinline__ float hw(const float* s, int b, int fy, int fx, int q) {
    return s[(((size_t)b * HYPER + q) * 8 + fy) * 8 + fx];
}

// ---------------------------------------------------------------------------
__global__ void k_zero(float* p, int n) {
    int i = blockIdx.x * blockDim.x + threadIdx.x;
    if (i < n) p[i] = 0.0f;
}

// ---------------------------------------------------------------------------
// Stage 1: x1[o][m] = sum_c patch[m][c] * w1[o][c]   (channel-major out)
// One block / patch, 8 waves. tn-outer (B fragment hoisted), tm-inner.
// Abuf zero-padded to 336 rows -> unconditional b128 fragment loads,
// padded rows compute to exact zeros -> unconditional stores & stats.
// ---------------------------------------------------------------------------
__global__ __launch_bounds__(256) void k_gemm1(
    const float* __restrict__ x, const float* __restrict__ s,
    bf16_t* __restrict__ x1c, float* __restrict__ sum1, float* __restrict__ sq1)
{
    __shared__ __align__(16) bf16_t   Abuf[M1P * IN_NC];   // 21504 B, MxK + zero pad
    __shared__ __align__(16) unsigned B1pT[HIDDEN * 16];   // 12288 B, pairs [o][k/2]
    __shared__ float ssum[HIDDEN], ssq[HIDDEN];

    const int p = blockIdx.x;
    const int b = p >> 6, fy = (p >> 3) & 7, fx = p & 7;
    const int tid = threadIdx.x;

    if (tid < HIDDEN) { ssum[tid] = 0.0f; ssq[tid] = 0.0f; }

    // stage patch (reflect halo), coalesced over m; zero the 12 pad rows
    for (int c = 0; c < IN_NC; ++c) {
        const float* xrow = x + (((size_t)b * IN_NC + c) << 14);
        for (int m = tid; m < M1; m += 256) {
            int i = m / 18, j = m - i * 18;
            int gy = fy * 16 + i - 1; gy = gy < 0 ? 1 : (gy > 127 ? 126 : gy);
            int gx = fx * 16 + j - 1; gx = gx < 0 ? 1 : (gx > 127 ? 126 : gx);
            Abuf[m * IN_NC + c] = (bf16_t)xrow[(gy << 7) + gx];
        }
    }
    for (int idx = tid; idx < (M1P - M1) * IN_NC; idx += 256)
        Abuf[M1 * IN_NC + idx] = (bf16_t)0.0f;

    // stage w1 as packed bf16 pairs, transposed [o][k/2] -> contiguous frags
    for (int idx = tid; idx < HIDDEN * 16; idx += 256) {
        int o = idx >> 4, k2 = idx & 15;
        int q = o * IN_NC + 2 * k2;
        Pair pk;
        pk.h[0] = (bf16_t)hw(s, b, fy, fx, q);
        pk.h[1] = (bf16_t)hw(s, b, fy, fx, q + 1);
        B1pT[idx] = pk.u;
    }
    __syncthreads();

    const int lane = tid & 31;
    const int wave = __builtin_amdgcn_readfirstlane(tid >> 5);  // SGPR: scalar loops
    const int half = lane >> 4, ml = lane & 15;

    for (int tn = 0; tn < 12; ++tn) {
        const int o = tn * 16 + ml;
        Frag bfr;                                   // invariant over tm: hoisted
        bfr.u4[0] = *(const uint4v*)&B1pT[o * 16 + half * 4];
        bfr.u4[1] = *(const uint4v*)&B1pT[o * 16 + 8 + half * 4];

        float s0 = 0.0f, s1 = 0.0f;                 // per-lane stats, channel o
        const unsigned* arow = (const unsigned*)Abuf + ((size_t)(wave * 16 + ml)) * 16;
        bf16_t* outp = x1c + ((size_t)p * HIDDEN + o) * M1P + wave * 16 + 8 * half;

        for (int tm = wave; tm < 21; tm += 8) {     // scalar-uniform trip
            Frag afr;                               // two ds_load_b128
            afr.u4[0] = *(const uint4v*)(arow + half * 4);
            afr.u4[1] = *(const uint4v*)(arow + 8 + half * 4);

            v8f acc = {};
            acc = __builtin_amdgcn_wmma_f32_16x16x32_bf16(
                false, afr.h, false, bfr.h, (short)0, acc, false, false);

            Pack8 pk;
            #pragma unroll
            for (int r = 0; r < 8; ++r) {
                float v = acc[r];
                pk.h[r] = (bf16_t)v;
                s0 += v; s1 += v * v;               // padded rows add exact 0
            }
            // D layout: lane%16=N, vgpr r -> M = r + 8*half : 8 consecutive M
            *(uint4v*)outp = pk.u4;
            outp += 128;                            // tm += 8 -> 128 elements
            arow += 128 * 16;                       // 128 rows of 16 dwords
        }
        atomicAdd(&ssum[o], s0);
        atomicAdd(&ssq[o], s1);
    }
    __syncthreads();
    if (tid < HIDDEN) { atomicAdd(&sum1[tid], ssum[tid]); atomicAdd(&sq1[tid], ssq[tid]); }
}

// ---------------------------------------------------------------------------
__global__ void k_bnfin(const float* __restrict__ sum, const float* __restrict__ sq,
                        const float* __restrict__ gamma, const float* __restrict__ beta,
                        float* __restrict__ scale, float* __restrict__ shift,
                        int nch, float invN)
{
    int c = blockIdx.x * blockDim.x + threadIdx.x;
    if (c < nch) {
        float m  = sum[c] * invN;
        float v  = sq[c] * invN - m * m;
        float sc = rsqrtf(v + 1e-5f) * gamma[c];
        scale[c] = sc;
        shift[c] = beta[c] - m * sc;
    }
}

// ---------------------------------------------------------------------------
// Stage 2: depthwise 3x3, BN1+ReLU6 fused on load. Channel loop is uniform:
// weights broadcast, stats reduced per-wave via shfl (wave32) then 1 atomic.
// ---------------------------------------------------------------------------
__global__ __launch_bounds__(256) void k_dwconv(
    const bf16_t* __restrict__ x1c, const float* __restrict__ s,
    const float* __restrict__ scale1, const float* __restrict__ shift1,
    bf16_t* __restrict__ x2c, float* __restrict__ sum2, float* __restrict__ sq2)
{
    __shared__ float w2l[HIDDEN * 9];
    __shared__ float sc[HIDDEN], sh[HIDDEN];
    __shared__ float ssum[HIDDEN], ssq[HIDDEN];

    const int p = blockIdx.x;
    const int b = p >> 6, fy = (p >> 3) & 7, fx = p & 7;
    const int tid = threadIdx.x;

    if (tid < HIDDEN) {
        sc[tid] = scale1[tid]; sh[tid] = shift1[tid];
        ssum[tid] = 0.0f; ssq[tid] = 0.0f;
    }
    for (int idx = tid; idx < HIDDEN * 9; idx += 256)
        w2l[idx] = hw(s, b, fy, fx, R1 + idx);
    __syncthreads();

    const bf16_t* xc = x1c + (size_t)p * HIDDEN * M1P;
    const int ij = tid, i = ij >> 4, j = ij & 15;

    for (int c = 0; c < HIDDEN; ++c) {              // uniform channel loop
        const bf16_t* row = xc + c * M1P;
        const float scc = sc[c], shc = sh[c];
        float acc = 0.0f;
        #pragma unroll
        for (int u = 0; u < 3; ++u)
            #pragma unroll
            for (int v = 0; v < 3; ++v) {
                float xv = (float)row[(i + u) * 18 + (j + v)];
                float h  = fminf(fmaxf(xv * scc + shc, 0.0f), 6.0f);
                acc += h * w2l[c * 9 + u * 3 + v];
            }
        x2c[((size_t)p * HIDDEN + c) * M3 + ij] = (bf16_t)acc;

        float rs = acc, rq = acc * acc;             // wave32 tree reduction
        #pragma unroll
        for (int off = 16; off > 0; off >>= 1) {
            rs += __shfl_down(rs, off, 32);
            rq += __shfl_down(rq, off, 32);
        }
        if ((tid & 31) == 0) { atomicAdd(&ssum[c], rs); atomicAdd(&ssq[c], rq); }
    }
    __syncthreads();
    if (tid < HIDDEN) { atomicAdd(&sum2[tid], ssum[tid]); atomicAdd(&sq2[tid], ssq[tid]); }
}

// ---------------------------------------------------------------------------
// Stage 3: x3[o][m] = sum_c relu6(bn2(x2[c][m])) * w3[o][c]
// Two 128-row phases (bounds LDS at ~62KB); 6-step K chain per tile.
// ---------------------------------------------------------------------------
__global__ __launch_bounds__(256) void k_gemm3(
    const bf16_t* __restrict__ x2c, const float* __restrict__ s,
    const float* __restrict__ scale2, const float* __restrict__ shift2,
    float* __restrict__ x3c, float* __restrict__ sum3, float* __restrict__ sq3)
{
    __shared__ __align__(16) bf16_t   A2[128 * HIDDEN];   // 49152 B (one M phase)
    __shared__ __align__(16) unsigned B3pT[OUT_NC * 96];  // 12288 B, pairs [o][k/2]
    __shared__ float sc[HIDDEN], sh[HIDDEN];
    __shared__ float ssum[OUT_NC], ssq[OUT_NC];

    const int p = blockIdx.x;
    const int b = p >> 6, fy = (p >> 3) & 7, fx = p & 7;
    const int tid = threadIdx.x;

    if (tid < OUT_NC) { ssum[tid] = 0.0f; ssq[tid] = 0.0f; }
    if (tid < HIDDEN) { sc[tid] = scale2[tid]; sh[tid] = shift2[tid]; }
    for (int idx = tid; idx < OUT_NC * 96; idx += 256) {
        int o = idx / 96, k2 = idx - o * 96;
        int q = R2 + o * HIDDEN + 2 * k2;
        Pair pk;
        pk.h[0] = (bf16_t)hw(s, b, fy, fx, q);
        pk.h[1] = (bf16_t)hw(s, b, fy, fx, q + 1);
        B3pT[idx] = pk.u;
    }

    const int lane = tid & 31;
    const int wave = __builtin_amdgcn_readfirstlane(tid >> 5);
    const int half = lane >> 4, nl = lane & 15;
    const int tml = wave >> 1, tn = wave & 1;       // one M-tile pair per wave
    const int o = tn * 16 + nl;

    for (int phase = 0; phase < 2; ++phase) {
        __syncthreads();                             // A2 reuse / B3pT first use
        for (int idx = tid; idx < 128 * HIDDEN; idx += 256) {
            int c = idx >> 7, r = idx & 127;         // coalesced over r
            float xv = (float)x2c[((size_t)p * HIDDEN + c) * M3 + phase * 128 + r];
            float h  = fminf(fmaxf(xv * sc[c] + sh[c], 0.0f), 6.0f);
            A2[r * HIDDEN + c] = (bf16_t)h;
        }
        __syncthreads();

        #pragma unroll
        for (int tt = 0; tt < 2; ++tt) {             // tiles wave and wave+8
            int tmlc = tml + tt * 4;                 // (wave+8)>>1 = tml+4
            int m = tmlc * 16 + nl;
            const unsigned* arow = (const unsigned*)&A2[m * HIDDEN];
            v8f acc = {};
            #pragma unroll
            for (int ks = 0; ks < 6; ++ks) {         // two b128 loads each side
                Frag afr, bfr;
                afr.u4[0] = *(const uint4v*)(arow + ks * 16 + half * 4);
                afr.u4[1] = *(const uint4v*)(arow + ks * 16 + 8 + half * 4);
                bfr.u4[0] = *(const uint4v*)&B3pT[o * 96 + ks * 16 + half * 4];
                bfr.u4[1] = *(const uint4v*)&B3pT[o * 96 + ks * 16 + 8 + half * 4];
                acc = __builtin_amdgcn_wmma_f32_16x16x32_bf16(
                    false, afr.h, false, bfr.h, (short)0, acc, false, false);
            }
            float s0 = 0.0f, s1 = 0.0f;
            #pragma unroll
            for (int r = 0; r < 8; ++r) { s0 += acc[r]; s1 += acc[r] * acc[r]; }
            Acc8 pk; pk.a = acc;
            size_t base = ((size_t)p * OUT_NC + o) * M3 + phase * 128 + tmlc * 16 + 8 * half;
            *(float4v*)&x3c[base]     = pk.q[0];
            *(float4v*)&x3c[base + 4] = pk.q[1];
            atomicAdd(&ssum[o], s0);
            atomicAdd(&ssq[o], s1);
        }
    }
    __syncthreads();
    if (tid < OUT_NC) { atomicAdd(&sum3[tid], ssum[tid]); atomicAdd(&sq3[tid], ssq[tid]); }
}

// ---------------------------------------------------------------------------
// BN3 + residual, un-patchify (x3 channel-major -> coalesced over j).
// ---------------------------------------------------------------------------
__global__ __launch_bounds__(256) void k_out(
    const float* __restrict__ x, const float* __restrict__ x3c,
    const float* __restrict__ scale3, const float* __restrict__ shift3,
    float* __restrict__ out)
{
    size_t t = (size_t)blockIdx.x * 256 + threadIdx.x;
    if (t >= (size_t)8 * OUT_NC * 128 * 128) return;
    int xi = (int)(t & 127);
    int y  = (int)((t >> 7) & 127);
    int oc = (int)((t >> 14) & 31);
    int b  = (int)(t >> 19);
    int fy = y >> 4, i = y & 15, fx = xi >> 4, j = xi & 15;
    int p  = b * 64 + fy * 8 + fx;
    float v = x3c[((size_t)p * OUT_NC + oc) * M3 + i * 16 + j];
    out[t] = x[t] + v * scale3[oc] + shift3[oc];
}

// ---------------------------------------------------------------------------
extern "C" void kernel_launch(void* const* d_in, const int* in_sizes, int n_in,
                              void* d_out, int out_size, void* d_ws, size_t ws_size,
                              hipStream_t stream) {
    const float* x  = (const float*)d_in[0];
    const float* s  = (const float*)d_in[1];
    const float* g1 = (const float*)d_in[2];
    const float* b1 = (const float*)d_in[3];
    const float* g2 = (const float*)d_in[4];
    const float* b2 = (const float*)d_in[5];
    const float* g3 = (const float*)d_in[6];
    const float* b3 = (const float*)d_in[7];
    float* out = (float*)d_out;

    // workspace layout (bytes)
    char* ws = (char*)d_ws;
    const size_t x1_bytes = (size_t)NPATCH * HIDDEN * M1P * sizeof(bf16_t); // 66,060,288
    const size_t x2_bytes = (size_t)NPATCH * HIDDEN * M3  * sizeof(bf16_t); // 50,331,648
    const size_t x3_bytes = (size_t)NPATCH * OUT_NC * M3  * sizeof(float);  // 16,777,216
    bf16_t* x1w = (bf16_t*)ws;
    bf16_t* x2w = (bf16_t*)(ws + x1_bytes);
    float*  x3w = (float*)(ws + x1_bytes + x2_bytes);
    float*  st  = (float*)(ws + x1_bytes + x2_bytes + x3_bytes);
    float *sum1 = st,        *sq1 = st + 192;
    float *sum2 = st + 384,  *sq2 = st + 576;
    float *sum3 = st + 768,  *sq3 = st + 800;
    float *sc1 = st + 832,  *sh1 = st + 1024;
    float *sc2 = st + 1216, *sh2 = st + 1408;
    float *sc3 = st + 1600, *sh3 = st + 1632;

    k_zero<<<4, 256, 0, stream>>>(st, 832);

    k_gemm1<<<NPATCH, 256, 0, stream>>>(x, s, x1w, sum1, sq1);
    k_bnfin<<<1, 192, 0, stream>>>(sum1, sq1, g1, b1, sc1, sh1, HIDDEN,
                                   1.0f / ((float)NPATCH * M1));

    k_dwconv<<<NPATCH, 256, 0, stream>>>(x1w, s, sc1, sh1, x2w, sum2, sq2);
    k_bnfin<<<1, 192, 0, stream>>>(sum2, sq2, g2, b2, sc2, sh2, HIDDEN,
                                   1.0f / ((float)NPATCH * M3));

    k_gemm3<<<NPATCH, 256, 0, stream>>>(x2w, s, sc2, sh2, x3w, sum3, sq3);
    k_bnfin<<<1, 32, 0, stream>>>(sum3, sq3, g3, b3, sc3, sh3, OUT_NC,
                                  1.0f / ((float)NPATCH * M3));

    const size_t n_out = (size_t)8 * OUT_NC * 128 * 128;
    k_out<<<(unsigned)((n_out + 255) / 256), 256, 0, stream>>>(x, x3w, sc3, sh3, out);
}
